// Sequence2SequenceDecoder_8040178778158
// MI455X (gfx1250) — matrix operand order repaired and checked by reading, not compiled, not run on hardware
//
#include <hip/hip_runtime.h>
#include <hip/hip_bf16.h>
#include <math.h>

typedef __attribute__((ext_vector_type(16))) _Float16 v16h;
typedef __attribute__((ext_vector_type(8)))  _Float16 v8h;
typedef __attribute__((ext_vector_type(8)))  float    v8f;

#define B_  128
#define S_  512
#define H_  1024
#define I_  256
#define G4  4096   // 4*H

// fast transcendentals: v_exp_f32 + v_rcp_f32, no IEEE divide sequence
__device__ __forceinline__ float fast_sigmoid(float x) {
    return __builtin_amdgcn_rcpf(1.0f + __expf(-x));
}
__device__ __forceinline__ float fast_tanh(float x) {
    // tanh(x) = 1 - 2/(e^{2x}+1); saturates correctly at +/-inf
    return 1.0f - 2.0f * __builtin_amdgcn_rcpf(1.0f + __expf(2.0f * x));
}

// ---------------------------------------------------------------------------
// Precompute: W_comb = W_hh + W_ih @ W_out   (fp32 accumulate, cast to f16)
// Also emit f16 copy of W_hh for step 0.
// ---------------------------------------------------------------------------
__global__ __launch_bounds__(256) void build_wcomb(
    const float* __restrict__ W_ih,   // [4096 x 256]
    const float* __restrict__ W_hh,   // [4096 x 1024]
    const float* __restrict__ W_out,  // [256 x 1024]
    _Float16* __restrict__ Wc,        // [4096 x 1024] f16
    _Float16* __restrict__ Whh16)     // [4096 x 1024] f16
{
    int idx = blockIdx.x * 256 + threadIdx.x;      // over 4096*1024
    int g = idx >> 10;
    int h = idx & 1023;
    float whh = W_hh[idx];
    float s = whh;
    const float* wi = W_ih + g * I_;
    #pragma unroll 4
    for (int k = 0; k < I_; ++k)
        s = fmaf(wi[k], W_out[k * H_ + h], s);
    Wc[idx]    = (_Float16)s;
    Whh16[idx] = (_Float16)whh;
}

// bias0 = b_ih + b_hh ; bias_comb = bias0 + W_ih @ b_out
__global__ __launch_bounds__(256) void build_bias(
    const float* __restrict__ b_ih, const float* __restrict__ b_hh,
    const float* __restrict__ W_ih, const float* __restrict__ b_out,
    float* __restrict__ bias0, float* __restrict__ biasc)
{
    int g = blockIdx.x * 256 + threadIdx.x;        // 4096 total
    float b0 = b_ih[g] + b_hh[g];
    float s = b0;
    const float* wi = W_ih + g * I_;
    #pragma unroll 4
    for (int k = 0; k < I_; ++k)
        s = fmaf(wi[k], b_out[k], s);
    bias0[g] = b0;
    biasc[g] = s;
}

// h0 (f16) <- encoder_output_h ; c <- 0
__global__ __launch_bounds__(256) void init_state(
    const float* __restrict__ enc, _Float16* __restrict__ h0,
    float* __restrict__ c)
{
    int idx = blockIdx.x * 256 + threadIdx.x;      // 128*1024
    h0[idx] = (_Float16)enc[idx];
    c[idx]  = 0.0f;
}

// ---------------------------------------------------------------------------
// One LSTM timestep, fully fused:
//   gates = h_in @ W^T + bias  (f16 WMMA, f32 acc)  -> LSTM elementwise
// Block: 256 threads (8 waves). Block tile: 16 batch rows x 32 h-columns.
// Wave w: gate type (w&3), column half (w>>2). Dual accumulator chains.
// grid = (H/32, B/16) = (32, 8)
// ---------------------------------------------------------------------------
__global__ __launch_bounds__(256) void lstm_step(
    const _Float16* __restrict__ W,    // [4096 x 1024] f16, row-major
    const float*    __restrict__ bias, // [4096]
    const _Float16* __restrict__ h_in, // [128 x 1024] f16
    float*          __restrict__ c,    // [128 x 1024] f32 (in/out)
    _Float16*       __restrict__ h_out,// [128 x 1024] f16
    float*          __restrict__ out,  // [128 x 512 x 1024] f32
    int t)
{
    // A tile: 16 rows x 1024 cols, stride padded to 1032 halfs (bank spread)
    __shared__ __align__(16) _Float16 shA[16 * 1032];
    __shared__ float shG[4 * 512];     // 4 gate tiles of 16x32 f32

    const int tid = threadIdx.x;
    const int n0  = blockIdx.x * 32;   // h-column tile base
    const int m0  = blockIdx.y * 16;   // batch tile base

    // Cooperative load of the h tile (16 x 1024 f16) into LDS, 16B chunks.
    for (int chunk = tid; chunk < 2048; chunk += 256) {
        int r  = chunk >> 7;           // 0..15
        int c8 = chunk & 127;          // 0..127 (chunks of 8 halfs)
        *(v8h*)&shA[r * 1032 + c8 * 8] =
            *(const v8h*)&h_in[(m0 + r) * H_ + c8 * 8];
    }
    __syncthreads();

    const int lane  = tid & 31;
    const int w     = tid >> 5;        // 0..7
    const int gate  = w & 3;           // i/f/g/o
    const int nh    = (w >> 2) << 4;   // 0 or 16: column half within tile
    const int khalf = lane >> 4;       // 0 or 1 (K half selector)
    const int ln    = lane & 15;       // M (for A) / N (for B)
    const int gr    = gate * 1024 + n0 + nh;  // first W row for this wave

    float bv = bias[gr + ln];
    v8f acc0 = {bv, bv, bv, bv, bv, bv, bv, bv};
    v8f acc1 = {0.f, 0.f, 0.f, 0.f, 0.f, 0.f, 0.f, 0.f};

    // B: lane ln = column N -> W row (gr+ln); lanes 0-15 K=0..15, 16-31 K=16..31
    const _Float16* wrow = W + (size_t)(gr + ln) * H_ + khalf * 16;
    // A: lane ln = row M; khalf=0 -> K 0..15, khalf=1 -> K 16..31
    const _Float16* arow = &shA[ln * 1032 + khalf * 16];

    // Two independent accumulation chains (even/odd 32-wide K slabs)
    #pragma unroll 2
    for (int k0 = 0; k0 < H_; k0 += 64) {
        union { v16h v; v8h h[2]; } a0, a1;
        a0.h[0] = *(const v8h*)&arow[k0];
        a0.h[1] = *(const v8h*)&arow[k0 + 8];
        v16h b0 = *(const v16h*)&wrow[k0];
        a1.h[0] = *(const v8h*)&arow[k0 + 32];
        a1.h[1] = *(const v8h*)&arow[k0 + 40];
        v16h b1 = *(const v16h*)&wrow[k0 + 32];
        acc0 = __builtin_amdgcn_wmma_f32_16x16x32_f16(
            false, a0.v, false, b0, (short)0, acc0, false, false);
        acc1 = __builtin_amdgcn_wmma_f32_16x16x32_f16(
            false, a1.v, false, b1, (short)0, acc1, false, false);
    }
    v8f acc = acc0 + acc1;

    // D layout: VGPR j, lane l -> M = j + 8*(l>>4), N = l&15
    #pragma unroll
    for (int j = 0; j < 8; ++j)
        shG[gate * 512 + (j + 8 * khalf) * 32 + nh + ln] = acc[j];
    __syncthreads();

    // Fused LSTM elementwise over the 16x32 tile (2 elements per thread)
    for (int e = tid; e < 512; e += 256) {
        int m = e >> 5, n = e & 31;
        float gi = shG[0 * 512 + e];
        float gf = shG[1 * 512 + e];
        float gg = shG[2 * 512 + e];
        float go = shG[3 * 512 + e];
        float i_ = fast_sigmoid(gi);
        float f_ = fast_sigmoid(gf);
        float g_ = fast_tanh(gg);
        float o_ = fast_sigmoid(go);
        int b  = m0 + m;
        int hc = n0 + n;
        float cp = c[b * H_ + hc];
        float cn = f_ * cp + i_ * g_;
        float hn = o_ * fast_tanh(cn);
        c[b * H_ + hc]     = cn;
        h_out[b * H_ + hc] = (_Float16)hn;
        // time-reversed output: step t -> out[b][S-1-t][:]
        out[((size_t)b * S_ + (S_ - 1 - t)) * H_ + hc] = hn;
    }
}

// ---------------------------------------------------------------------------
// Host launcher
// ---------------------------------------------------------------------------
extern "C" void kernel_launch(void* const* d_in, const int* in_sizes, int n_in,
                              void* d_out, int out_size, void* d_ws, size_t ws_size,
                              hipStream_t stream) {
    // inputs: x, encoder_output_h, W_ih, W_hh, b_ih, b_hh, W_out, b_out
    const float* enc   = (const float*)d_in[1];
    const float* W_ih  = (const float*)d_in[2];
    const float* W_hh  = (const float*)d_in[3];
    const float* b_ih  = (const float*)d_in[4];
    const float* b_hh  = (const float*)d_in[5];
    const float* W_out = (const float*)d_in[6];
    const float* b_out = (const float*)d_in[7];
    float* out = (float*)d_out;

    char* ws = (char*)d_ws;
    _Float16* Wc    = (_Float16*)(ws);                      //  8 MB
    _Float16* Whh16 = (_Float16*)(ws + 8388608);            //  8 MB
    float*    bias0 = (float*)   (ws + 16777216);           // 16 KB
    float*    biasc = (float*)   (ws + 16793600);           // 16 KB
    _Float16* hb0   = (_Float16*)(ws + 16809984);           // 256 KB
    _Float16* hb1   = (_Float16*)(ws + 17072128);           // 256 KB
    float*    cbuf  = (float*)   (ws + 17334272);           // 512 KB

    build_wcomb<<<(G4 * H_) / 256, 256, 0, stream>>>(W_ih, W_hh, W_out, Wc, Whh16);
    build_bias<<<G4 / 256, 256, 0, stream>>>(b_ih, b_hh, W_ih, b_out, bias0, biasc);
    init_state<<<(B_ * H_) / 256, 256, 0, stream>>>(enc, hb0, cbuf);

    dim3 grid(H_ / 32, B_ / 16);   // (32, 8)
    _Float16* hbuf[2] = {hb0, hb1};

    // Step 0: inp == 0 -> use plain W_hh (f16) and b_ih+b_hh
    lstm_step<<<grid, 256, 0, stream>>>(Whh16, bias0, hb0, cbuf, hb1, out, 0);
    // Steps 1..511: folded weights W_comb = W_hh + W_ih@W_out
    for (int t = 1; t < S_; ++t) {
        lstm_step<<<grid, 256, 0, stream>>>(Wc, biasc,
                                            hbuf[t & 1], cbuf,
                                            hbuf[(t + 1) & 1], out, t);
    }
}
